// PartCEMV0_7894149890694
// MI455X (gfx1250) — matrix operand order repaired and checked by reading, not compiled, not run on hardware
//
#include <hip/hip_runtime.h>
#include <stdint.h>

// Problem constants (match reference setup_inputs()).
#define BB   32
#define CC   2048
#define HH   14
#define WW   14
#define H2   28
#define W2   28
#define NN   784            // H2*W2
#define KK   113
#define KM1  112
#define NCLS 200
#define NT   49             // NN / 16 column tiles
#define KPAD 128            // KK padded to 8 row-tiles of 16
#define CP   (CC / 2)       // bf16 pairs per row
#define KT   64             // K-steps of 32 (CC / 32)

typedef __attribute__((ext_vector_type(16))) __bf16   v16bf;
typedef __attribute__((ext_vector_type(8)))  float    v8f;
typedef __attribute__((ext_vector_type(8)))  unsigned v8u;

static __device__ __forceinline__ unsigned short f2bf(float f) {
  unsigned u = __float_as_uint(f);
  u += 0x7fffu + ((u >> 16) & 1u);      // round-to-nearest-even
  return (unsigned short)(u >> 16);
}

static __device__ __forceinline__ v16bf frag_cast(uint4 a, uint4 b) {
  v8u u = {a.x, a.y, a.z, a.w, b.x, b.y, b.z, b.w};
  return __builtin_bit_cast(v16bf, u);
}

// ---------------------------------------------------------------- K0a
// concepts -> pre-swizzled WMMA A-fragment buffer:
// Abuf[kt][rowtile][lane][j] (bf16 pairs), rows >= 113 zero-padded.
// Contiguous K order, matching B: lane<16 holds K 0-15 of row M,
// lane>=16 holds K 16-31.
__global__ __launch_bounds__(256) void k0_aswz(
    const float* __restrict__ concepts, unsigned* __restrict__ Ab) {
  int idx = blockIdx.x * 256 + threadIdx.x;       // < KT*8*32*8 = 131072
  int j    = idx & 7;
  int lane = (idx >> 3) & 31;
  int rt   = (idx >> 8) & 7;
  int kt   = idx >> 11;
  int half = lane >> 4, lrow = lane & 15;
  int row  = rt * 16 + lrow;
  int kp   = kt * 16 + j + (half << 3);
  unsigned v = 0u;
  if (row < KK) {
    float f0 = concepts[(size_t)row * CC + 2 * kp];
    float f1 = concepts[(size_t)row * CC + 2 * kp + 1];
    v = (unsigned)f2bf(f0) | ((unsigned)f2bf(f1) << 16);
  }
  Ab[idx] = v;
}

// ---------------------------------------------------------------- K0b
// per-concept squared norm (deterministic tree reduce)
__global__ __launch_bounds__(256) void k0_csq(
    const float* __restrict__ concepts, float* __restrict__ csq) {
  __shared__ float red[256];
  int row = blockIdx.x, t = threadIdx.x;
  float s = 0.f;
  for (int c = t; c < CC; c += 256) {
    float v = concepts[(size_t)row * CC + c];
    s += v * v;
  }
  red[t] = s; __syncthreads();
  for (int o = 128; o > 0; o >>= 1) { if (t < o) red[t] += red[t + o]; __syncthreads(); }
  if (t == 0) csq[row] = red[0];
}

// ---------------------------------------------------------------- K1
// Fused bilinear 2x upsample (align_corners=False) writing bf16 pairs
// directly in WMMA B-fragment layout Bbuf[b][ntile][kt][lane][j],
// plus fp32 x_sq[b][n] and proj[b][n] = xe[b][n] . concept_fc_w.
// B layout: lane<16 holds K 0-15 of col N=lane, lane>=16 holds K 16-31.
__global__ __launch_bounds__(256) void k1_upsample(
    const float* __restrict__ x, const float* __restrict__ wfc,
    unsigned* __restrict__ Bb, float* __restrict__ xsq,
    float* __restrict__ proj) {
  __shared__ float r0[256], r1[256];
  int n = blockIdx.x, b = blockIdx.y, t = threadIdx.x;
  int ntile = n >> 4, lrow = n & 15;
  int oy = n / W2, ox = n % W2;
  float sy = oy * 0.5f - 0.25f, sx = ox * 0.5f - 0.25f;
  float fy = floorf(sy), fx = floorf(sx);
  float wy1 = sy - fy, wx1 = sx - fx;
  int y0 = max(0, min(HH - 1, (int)fy)), y1 = max(0, min(HH - 1, (int)fy + 1));
  int x0 = max(0, min(WW - 1, (int)fx)), x1 = max(0, min(WW - 1, (int)fx + 1));
  float w00 = (1.f - wy1) * (1.f - wx1), w01 = (1.f - wy1) * wx1;
  float w10 = wy1 * (1.f - wx1),         w11 = wy1 * wx1;
  int i00 = y0 * WW + x0, i01 = y0 * WW + x1;
  int i10 = y1 * WW + x0, i11 = y1 * WW + x1;
  float ssq = 0.f, sp = 0.f;
  const float* xb = x + (size_t)b * CC * HH * WW;
  unsigned* Bt = Bb + (((size_t)b * NT + ntile) * KT) * 256;  // this (b,ntile)
  for (int p = t; p < CP; p += 256) {     // 1024 bf16 pairs (c = 2p, 2p+1)
    const float* p0 = xb + (size_t)(2 * p) * (HH * WW);
    const float* p1 = p0 + HH * WW;
    float v0 = w00 * p0[i00] + w01 * p0[i01] + w10 * p0[i10] + w11 * p0[i11];
    float v1 = w00 * p1[i00] + w01 * p1[i01] + w10 * p1[i10] + w11 * p1[i11];
    unsigned u = (unsigned)f2bf(v0) | ((unsigned)f2bf(v1) << 16);
    int kt = p >> 4, jj = p & 15;
    int half = jj >> 3, j = jj & 7;
    int lane = (half << 4) | lrow;
    Bt[((size_t)kt * 32 + lane) * 8 + j] = u;
    ssq += v0 * v0 + v1 * v1;
    sp  += v0 * wfc[2 * p] + v1 * wfc[2 * p + 1];
  }
  r0[t] = ssq; r1[t] = sp; __syncthreads();
  for (int o = 128; o > 0; o >>= 1) {
    if (t < o) { r0[t] += r0[t + o]; r1[t] += r1[t + o]; }
    __syncthreads();
  }
  if (t == 0) { xsq[b * NN + n] = r0[0]; proj[b * NN + n] = r1[0]; }
}

// ---------------------------------------------------------------- K2
// Per (n-tile, batch-pair) workgroup: 8 waves = 8 concept row-tiles.
// Main loop: direct-to-register b128 fragment loads (no LDS, no barriers),
// 2 WMMAs per wave per K-step (two batches share the A fragment).
// Epilogue: fused sqrt-distance, softmax over 113 concepts, score_maps
// write, and per-tile logit partials (attn[:112] . proj) via LDS.
__global__ __launch_bounds__(256) void k2_dots_softmax(
    const uint4* __restrict__ Ab,       // [KT][8][32][2] uint4
    const uint4* __restrict__ Bb,       // [B][NT][KT][32][2] uint4
    const float* __restrict__ csq, const float* __restrict__ xsq,
    const float* __restrict__ proj,
    float* __restrict__ out_maps,       // [B][K][N]
    float* __restrict__ part) {         // [B][NT][KM1]
  __shared__ float distS[2][KPAD][17];
  __shared__ float projS[2][16], xsqS[2][16], csqS[KPAD];

  int ntile = blockIdx.x, bpair = blockIdx.y, t = threadIdx.x;
  int b0 = bpair * 2, n0 = ntile * 16;
  if (t < 32) {
    int bs = t >> 4, col = t & 15;
    projS[bs][col] = proj[(b0 + bs) * NN + n0 + col];
    xsqS[bs][col]  = xsq[(b0 + bs) * NN + n0 + col];
  }
  if (t < KPAD) csqS[t] = (t < KK) ? csq[t] : 0.f;

  int lane = t & 31, wv = t >> 5;       // wv = concept row tile (0..7)
  int half = lane >> 4, lrow = lane & 15;
  v8f acc0 = {}, acc1 = {};

  const uint4* pA  = Ab + ((size_t)wv * 32 + lane) * 2;
  const uint4* pB0 = Bb + (((size_t)b0 * NT + ntile) * KT * 32 + lane) * 2;
  const uint4* pB1 = pB0 + (size_t)NT * KT * 32 * 2;   // next batch
#pragma unroll 2
  for (int kt = 0; kt < KT; ++kt) {
    uint4 a0 = pA[0],  a1 = pA[1];
    uint4 c0 = pB0[0], c1 = pB0[1];
    uint4 d0 = pB1[0], d1 = pB1[1];
    v16bf av  = frag_cast(a0, a1);
    v16bf bv0 = frag_cast(c0, c1);
    v16bf bv1 = frag_cast(d0, d1);
    acc0 = __builtin_amdgcn_wmma_f32_16x16x32_bf16(
        false, av, false, bv0, (short)0, acc0, false, false);
    acc1 = __builtin_amdgcn_wmma_f32_16x16x32_bf16(
        false, av, false, bv1, (short)0, acc1, false, false);
    pA  += 8 * 32 * 2;                  // next K-step
    pB0 += 32 * 2;
    pB1 += 32 * 2;
  }
  __syncthreads();                      // csqS/xsqS/projS visible

  // D layout: acc[r] -> row = 16*wv + r + 8*half, col = lrow
#pragma unroll
  for (int r = 0; r < 8; ++r) {
    int row = (wv << 4) + r + (half << 3);
    float cs = csqS[row];
    distS[0][row][lrow] = sqrtf(fmaxf(cs + xsqS[0][lrow] - 2.f * acc0[r], 0.f));
    distS[1][row][lrow] = sqrtf(fmaxf(cs + xsqS[1][lrow] - 2.f * acc1[r], 0.f));
  }
  __syncthreads();

  if (t < 32) {                         // stable softmax over k per column
    int bs = t >> 4, col = t & 15;
    float m = distS[bs][0][col];
    for (int k = 1; k < KK; ++k) m = fminf(m, distS[bs][k][col]);
    float s = 0.f;
    for (int k = 0; k < KK; ++k) s += __expf(m - distS[bs][k][col]);
    float inv = 1.f / s;
    for (int k = 0; k < KK; ++k) distS[bs][k][col] = __expf(m - distS[bs][k][col]) * inv;
  }
  __syncthreads();

  for (int bs = 0; bs < 2; ++bs)        // score_maps (== attn)
    for (int i = t; i < KK * 16; i += 256) {
      int k = i >> 4, col = i & 15;
      out_maps[((size_t)(b0 + bs) * KK + k) * NN + n0 + col] = distS[bs][k][col];
    }

  int bs = t >> 7, k = t & 127;         // per-tile logit partials (no atomics)
  if (k < KM1) {
    float s = 0.f;
#pragma unroll
    for (int col = 0; col < 16; ++col) s += distS[bs][k][col] * projS[bs][col];
    part[((size_t)(b0 + bs) * NT + ntile) * KM1 + k] = s;
  }
}

// ---------------------------------------------------------------- K3
__global__ __launch_bounds__(128) void k3_logits(
    const float* __restrict__ part, const float* __restrict__ cfb,
    float* __restrict__ out_scores, float* __restrict__ scores_ws) {
  int b = blockIdx.x, k = threadIdx.x;
  if (k >= KM1) return;
  float s = 0.f;
  for (int tl = 0; tl < NT; ++tl) s += part[((size_t)b * NT + tl) * KM1 + k];
  s += cfb[0];
  float sc = 1.f / (1.f + __expf(-s));
  out_scores[b * KM1 + k] = sc;
  scores_ws[b * KM1 + k]  = sc;
}

// ---------------------------------------------------------------- K4
__global__ __launch_bounds__(256) void k4_tmp(
    const float* __restrict__ scores, const float* __restrict__ concepts,
    float* __restrict__ tmp) {
  int c = blockIdx.x * 256 + threadIdx.x;
  int b = blockIdx.y;
  float s = 0.f;
  for (int k = 0; k < KM1; ++k) s += scores[b * KM1 + k] * concepts[(size_t)k * CC + c];
  tmp[(size_t)b * CC + c] = s;
}

// ---------------------------------------------------------------- K5
__global__ __launch_bounds__(256) void k5_preds(
    const float* __restrict__ tmp, const float* __restrict__ lfw,
    const float* __restrict__ lfb, float* __restrict__ out_preds) {
  int gw = blockIdx.x * 8 + (threadIdx.x >> 5);   // one wave32 per (b, class)
  int lane = threadIdx.x & 31;
  if (gw >= BB * NCLS) return;
  int b = gw / NCLS, j = gw % NCLS;
  float s = 0.f;
  for (int c = lane; c < CC; c += 32)
    s += tmp[(size_t)b * CC + c] * lfw[(size_t)j * CC + c];
  for (int o = 16; o > 0; o >>= 1) s += __shfl_xor(s, o, 32);
  if (lane == 0) out_preds[b * NCLS + j] = s + lfb[j];
}

// ----------------------------------------------------------------
extern "C" void kernel_launch(void* const* d_in, const int* in_sizes, int n_in,
                              void* d_out, int out_size, void* d_ws, size_t ws_size,
                              hipStream_t stream) {
  (void)in_sizes; (void)n_in; (void)out_size; (void)ws_size;
  const float* x        = (const float*)d_in[0];
  const float* concepts = (const float*)d_in[1];
  const float* cfw      = (const float*)d_in[2];
  const float* cfb      = (const float*)d_in[3];
  const float* lfw      = (const float*)d_in[4];
  const float* lfb      = (const float*)d_in[5];

  float* out        = (float*)d_out;
  float* out_maps   = out;                                   // B*K*N
  float* out_scores = out + (size_t)BB * KK * NN;            // B*112
  float* out_preds  = out_scores + (size_t)BB * KM1;         // B*200

  char* ws = (char*)d_ws;
  size_t off = 0;
  auto wsalloc = [&](size_t bytes) -> void* {
    void* p = ws + off;
    off += (bytes + 255) & ~(size_t)255;
    return p;
  };
  unsigned* Ab  = (unsigned*)wsalloc((size_t)KT * 8 * 32 * 8 * 4);           // 512 KB
  unsigned* Bbf = (unsigned*)wsalloc((size_t)BB * NT * KT * 32 * 8 * 4);     // ~98 MB
  float* csq    = (float*)wsalloc(KK * sizeof(float));
  float* xsq    = (float*)wsalloc((size_t)BB * NN * sizeof(float));
  float* proj   = (float*)wsalloc((size_t)BB * NN * sizeof(float));
  float* part   = (float*)wsalloc((size_t)BB * NT * KM1 * sizeof(float));
  float* scores = (float*)wsalloc((size_t)BB * KM1 * sizeof(float));
  float* tmp    = (float*)wsalloc((size_t)BB * CC * sizeof(float));

  k0_aswz<<<dim3((KT * 8 * 32 * 8) / 256), dim3(256), 0, stream>>>(concepts, Ab);
  k0_csq<<<dim3(KK), dim3(256), 0, stream>>>(concepts, csq);
  k1_upsample<<<dim3(NN, BB), dim3(256), 0, stream>>>(x, cfw, Bbf, xsq, proj);
  k2_dots_softmax<<<dim3(NT, BB / 2), dim3(256), 0, stream>>>(
      (const uint4*)Ab, (const uint4*)Bbf, csq, xsq, proj, out_maps, part);
  k3_logits<<<dim3(BB), dim3(128), 0, stream>>>(part, cfb, out_scores, scores);
  k4_tmp<<<dim3(CC / 256, BB), dim3(256), 0, stream>>>(scores, concepts, tmp);
  k5_preds<<<dim3((BB * NCLS + 7) / 8), dim3(256), 0, stream>>>(tmp, lfw, lfb, out_preds);
}
